// GraphSAGE_26319559590589
// MI455X (gfx1250) — matrix operand order repaired and checked by reading, not compiled, not run on hardware
//
#include <hip/hip_runtime.h>

typedef __attribute__((ext_vector_type(16))) _Float16 v16h;
typedef __attribute__((ext_vector_type(8)))  _Float16 v8h;
typedef __attribute__((ext_vector_type(8)))  float    v8f;

#define NEG_SLOPE 0.01f

// ---------------- problem constants (from setup_inputs) ----------------
static const int N_DST0 = 65536;   // layer-1 dst nodes
static const int N_DST1 = 4096;    // layer-2 dst nodes
static const int F_IN   = 128;
static const int F_HID  = 256;
static const int F_OUT  = 32;
static const int FANOUT = 16;

// ---------------- ws layout (bytes) ----------------
static const size_t OFF_SCAL = 0;                       // [0]=normsum0 [1]=mean0 [2]=normsum1 [3]=mean1
static const size_t OFF_W1N  = 256;                     // 128x256 f16 frag-packed = 65536 B
static const size_t OFF_W1S  = OFF_W1N + 65536;
static const size_t OFF_W2N  = OFF_W1S + 65536;         // 256x32 f16 frag-packed = 16384 B
static const size_t OFF_W2S  = OFF_W2N + 16384;
static const size_t OFF_H    = OFF_W2S + 16384;         // h: 65536x256 f16 = 33554432 B

// ======================================================================
// Fragment loaders per CDNA5 WMMA VGPR layouts (ISA 7.12.2, wave32).
// A 16x32 f16 from row-major LDS tile (ldk = row stride in halfs):
//   lane<16 : row=lane,    halfs 0..7 -> K=kt*32+0..7,  8..15 -> K=kt*32+16..23
//   lane>=16: row=lane-16, halfs 0..7 -> K=kt*32+8..15, 8..15 -> K=kt*32+24..31
__device__ __forceinline__ v16h load_a_frag(const _Float16* A, int ldk, int kt, int lane) {
    int row = lane & 15;
    int k0  = kt * 32 + ((lane >> 4) << 3);
    v8h lo = *(const v8h*)(A + row * ldk + k0);
    v8h hi = *(const v8h*)(A + row * ldk + k0 + 16);
    return __builtin_shufflevector(lo, hi, 0,1,2,3,4,5,6,7,8,9,10,11,12,13,14,15);
}
// B 32x16 f16 pre-packed fragment-major: frag = nt*nKt + kt; per lane 16 contiguous halfs.
__device__ __forceinline__ v16h load_b_frag(const _Float16* P, int frag, int lane) {
    return *(const v16h*)(P + (size_t)frag * 512 + lane * 16);
}
__device__ __forceinline__ v8f wmma_f16(v16h a, v16h b, v8f c) {
    return __builtin_amdgcn_wmma_f32_16x16x32_f16(false, a, false, b, (short)0, c, false, false);
}

// ======================================================================
// Pack W (Ktot x Ntot, row-major f32) -> fragment-major f16.
// frag = nt*nKt + kt ; within frag: [lane][i] with
//   K = kt*32 + (lane>=16 ? 8 : 0) + (i&7) + (i>=8 ? 16 : 0) ,  N = nt*16 + (lane&15)
__global__ void k_pack_w(const float* __restrict__ W, _Float16* __restrict__ out,
                         int Ntot, int nKt, int totalHalfs) {
    int tid = blockIdx.x * blockDim.x + threadIdx.x;
    if (tid >= totalHalfs) return;
    int i    = tid & 15;
    int lane = (tid >> 4) & 31;
    int frag = tid >> 9;
    int kt = frag % nKt;
    int nt = frag / nKt;
    int K = kt * 32 + ((lane >> 4) << 3) + (i & 7) + ((i >> 3) << 4);
    int N = nt * 16 + (lane & 15);
    out[tid] = (_Float16)W[(size_t)K * Ntot + N];
}

// ---------------- row-norm sum (f32 rows, cols=128): 8 rows/block ----------------
__global__ void k_normsum_f32(const float* __restrict__ x, float* __restrict__ accum) {
    int row  = blockIdx.x * 8 + (threadIdx.x >> 5);
    int lane = threadIdx.x & 31;
    const float* p = x + (size_t)row * F_IN + lane * 4;
    float s = p[0]*p[0] + p[1]*p[1] + p[2]*p[2] + p[3]*p[3];
    #pragma unroll
    for (int off = 16; off > 0; off >>= 1) s += __shfl_xor(s, off, 32);
    if (lane == 0) atomicAdd(accum, sqrtf(s));
}

// ---------------- row-norm sum (f16 rows, cols=256): 8 rows/block ----------------
__global__ void k_normsum_f16(const _Float16* __restrict__ h, float* __restrict__ accum) {
    int row  = blockIdx.x * 8 + (threadIdx.x >> 5);
    int lane = threadIdx.x & 31;
    v8h v = *(const v8h*)(h + (size_t)row * F_HID + lane * 8);
    float s = 0.f;
    #pragma unroll
    for (int k = 0; k < 8; ++k) { float f = (float)v[k]; s += f * f; }
    #pragma unroll
    for (int off = 16; off > 0; off >>= 1) s += __shfl_xor(s, off, 32);
    if (lane == 0) atomicAdd(accum, sqrtf(s));
}

__global__ void k_finalize_mean(const float* __restrict__ acc, float* __restrict__ mean, float inv) {
    *mean = *acc * inv;
}

// ======================================================================
// Layer 1: 16 dst rows x 256 cols per block (256 thr = 8 waves, 2 N-tiles/wave).
// Fused: gather/mean -> f16 LDS -> WMMA (neigh+self into one C) -> +b1 ->
//        row norm -> *size_b_mean -> leaky relu -> f16 h out.
__global__ void __launch_bounds__(256)
k_layer1(const float* __restrict__ x, const int* __restrict__ esrc,
         const _Float16* __restrict__ Wn, const _Float16* __restrict__ Ws,
         const float* __restrict__ b1, const float* __restrict__ mean0,
         _Float16* __restrict__ hout) {
    __shared__ _Float16 An[16][128];
    __shared__ _Float16 As[16][128];
    __shared__ float    HT[16][256];
    __shared__ float    pn[16][16];
    __shared__ float    rs[16];

    const int t = threadIdx.x;
    const int dstBase = blockIdx.x * 16;
    const int d = t >> 4, i = t & 15;

    // --- self features: x[dst] -> f16 ---
    {
        const float* src = x + (size_t)(dstBase + d) * F_IN + i * 8;
        float4 a = *(const float4*)src;
        float4 b = *(const float4*)(src + 4);
        _Float16* o = &As[d][i * 8];
        o[0]=(_Float16)a.x; o[1]=(_Float16)a.y; o[2]=(_Float16)a.z; o[3]=(_Float16)a.w;
        o[4]=(_Float16)b.x; o[5]=(_Float16)b.y; o[6]=(_Float16)b.z; o[7]=(_Float16)b.w;
    }
    // --- neighbor mean (deg == FANOUT, consecutive edges) ---
    {
        float acc[8] = {0,0,0,0,0,0,0,0};
        const int* e = esrc + (size_t)(dstBase + d) * FANOUT;
        for (int j = 0; j < FANOUT; ++j) {
            const float* src = x + (size_t)e[j] * F_IN + i * 8;
            float4 a = *(const float4*)src;
            float4 b = *(const float4*)(src + 4);
            acc[0]+=a.x; acc[1]+=a.y; acc[2]+=a.z; acc[3]+=a.w;
            acc[4]+=b.x; acc[5]+=b.y; acc[6]+=b.z; acc[7]+=b.w;
        }
        _Float16* o = &An[d][i * 8];
        #pragma unroll
        for (int k = 0; k < 8; ++k) o[k] = (_Float16)(acc[k] * (1.0f / FANOUT));
    }
    __syncthreads();

    // --- GEMM: wave w covers N-tiles 2w, 2w+1; K = 128 -> 4 steps of 32 ---
    const int wave = t >> 5, lane = t & 31;
    for (int nt = wave * 2; nt < wave * 2 + 2; ++nt) {
        v8f c = {};
        #pragma unroll
        for (int kt = 0; kt < 4; ++kt) {
            v16h a  = load_a_frag(&An[0][0], 128, kt, lane);
            v16h bm = load_b_frag(Wn, nt * 4 + kt, lane);
            c = wmma_f16(a, bm, c);
            a  = load_a_frag(&As[0][0], 128, kt, lane);
            bm = load_b_frag(Ws, nt * 4 + kt, lane);
            c = wmma_f16(a, bm, c);
        }
        const int m0 = (lane >= 16) ? 8 : 0;
        const int n  = (lane & 15) + nt * 16;
        const float bias = b1[n];
        #pragma unroll
        for (int r = 0; r < 8; ++r) HT[m0 + r][n] = c[r] + bias;
    }
    __syncthreads();

    // --- row norms ---
    {
        float s = 0.f;
        #pragma unroll
        for (int cidx = 0; cidx < 16; ++cidx) { float v = HT[d][i * 16 + cidx]; s += v * v; }
        pn[d][i] = s;
    }
    __syncthreads();
    if (t < 16) {
        float tot = 0.f;
        #pragma unroll
        for (int j = 0; j < 16; ++j) tot += pn[t][j];
        rs[t] = (*mean0) / sqrtf(tot);
    }
    __syncthreads();

    // --- rescale + leaky relu + store f16 ---
    {
        const float sc = rs[d];
        _Float16 tmp[16];
        #pragma unroll
        for (int cc = 0; cc < 16; ++cc) {
            float v = HT[d][i * 16 + cc] * sc;
            v = (v >= 0.f) ? v : NEG_SLOPE * v;
            tmp[cc] = (_Float16)v;
        }
        _Float16* o = hout + (size_t)(dstBase + d) * F_HID + i * 16;
        *(v8h*)o       = *(const v8h*)&tmp[0];
        *(v8h*)(o + 8) = *(const v8h*)&tmp[8];
    }
}

// ======================================================================
// Layer 2: 32 dst rows x 32 cols per block (128 thr = 4 waves, 1 tile/wave).
__global__ void __launch_bounds__(128)
k_layer2(const _Float16* __restrict__ h, const int* __restrict__ esrc,
         const _Float16* __restrict__ Wn, const _Float16* __restrict__ Ws,
         const float* __restrict__ b2, const float* __restrict__ mean1,
         float* __restrict__ out) {
    __shared__ _Float16 An[32][256];
    __shared__ _Float16 As[32][256];
    __shared__ float    HT[32][32];
    __shared__ float    pn[32][4];
    __shared__ float    rs[32];

    const int t = threadIdx.x;
    const int base = blockIdx.x * 32;
    const int i = t & 15;

    // --- gather/mean + self copy: 8 dsts per pass, 4 passes ---
    for (int dd = (t >> 4); dd < 32; dd += 8) {
        // self
        {
            const _Float16* src = h + (size_t)(base + dd) * F_HID + i * 16;
            *(v8h*)&As[dd][i * 16]     = *(const v8h*)src;
            *(v8h*)&As[dd][i * 16 + 8] = *(const v8h*)(src + 8);
        }
        // neighbor mean
        float acc[16];
        #pragma unroll
        for (int k = 0; k < 16; ++k) acc[k] = 0.f;
        const int* e = esrc + (size_t)(base + dd) * FANOUT;
        for (int j = 0; j < FANOUT; ++j) {
            const _Float16* src = h + (size_t)e[j] * F_HID + i * 16;
            v8h lo = *(const v8h*)src;
            v8h hi = *(const v8h*)(src + 8);
            #pragma unroll
            for (int k = 0; k < 8; ++k) { acc[k] += (float)lo[k]; acc[8 + k] += (float)hi[k]; }
        }
        #pragma unroll
        for (int k = 0; k < 16; ++k) An[dd][i * 16 + k] = (_Float16)(acc[k] * (1.0f / FANOUT));
    }
    __syncthreads();

    // --- GEMM: wave w -> row-group rg = w>>1, N-tile nt = w&1; K=256 -> 8 steps ---
    const int wave = t >> 5, lane = t & 31;
    const int rg = wave >> 1, nt = wave & 1;
    {
        v8f c = {};
        #pragma unroll
        for (int kt = 0; kt < 8; ++kt) {
            v16h a  = load_a_frag(&An[rg * 16][0], 256, kt, lane);
            v16h bm = load_b_frag(Wn, nt * 8 + kt, lane);
            c = wmma_f16(a, bm, c);
            a  = load_a_frag(&As[rg * 16][0], 256, kt, lane);
            bm = load_b_frag(Ws, nt * 8 + kt, lane);
            c = wmma_f16(a, bm, c);
        }
        const int m0 = rg * 16 + ((lane >= 16) ? 8 : 0);
        const int n  = (lane & 15) + nt * 16;
        const float bias = b2[n];
        #pragma unroll
        for (int r = 0; r < 8; ++r) HT[m0 + r][n] = c[r] + bias;
    }
    __syncthreads();

    // --- row norms (32 cols): 4 threads/row x 8 cols ---
    {
        const int d = t >> 2, ii = t & 3;
        float s = 0.f;
        #pragma unroll
        for (int cc = 0; cc < 8; ++cc) { float v = HT[d][ii * 8 + cc]; s += v * v; }
        pn[d][ii] = s;
    }
    __syncthreads();
    if (t < 32) {
        float tot = pn[t][0] + pn[t][1] + pn[t][2] + pn[t][3];
        rs[t] = (*mean1) / sqrtf(tot);
    }
    __syncthreads();
    {
        const int d = t >> 2, ii = t & 3;
        const float sc = rs[d];
        float* o = out + (size_t)(base + d) * F_OUT + ii * 8;
        #pragma unroll
        for (int cc = 0; cc < 8; ++cc) {
            float v = HT[d][ii * 8 + cc] * sc;
            o[cc] = (v >= 0.f) ? v : NEG_SLOPE * v;
        }
    }
}

// ======================================================================
extern "C" void kernel_launch(void* const* d_in, const int* in_sizes, int n_in,
                              void* d_out, int out_size, void* d_ws, size_t ws_size,
                              hipStream_t stream) {
    const float* x     = (const float*)d_in[0];
    const int*   es0   = (const int*)  d_in[1];
    const int*   es1   = (const int*)  d_in[3];
    const float* Wn1   = (const float*)d_in[5];
    const float* Ws1   = (const float*)d_in[6];
    const float* b1    = (const float*)d_in[7];
    const float* Wn2   = (const float*)d_in[8];
    const float* Ws2   = (const float*)d_in[9];
    const float* b2    = (const float*)d_in[10];
    float* out = (float*)d_out;

    char* ws = (char*)d_ws;
    float*     scal = (float*)(ws + OFF_SCAL);     // [0] sum0 [1] mean0 [2] sum1 [3] mean1
    _Float16*  w1n  = (_Float16*)(ws + OFF_W1N);
    _Float16*  w1s  = (_Float16*)(ws + OFF_W1S);
    _Float16*  w2n  = (_Float16*)(ws + OFF_W2N);
    _Float16*  w2s  = (_Float16*)(ws + OFF_W2S);
    _Float16*  hbuf = (_Float16*)(ws + OFF_H);

    hipMemsetAsync(scal, 0, 4 * sizeof(float), stream);

    // size_b_mean over x[:N_DST0] (runs before layer1 consumes mean0)
    k_normsum_f32<<<N_DST0 / 8, 256, 0, stream>>>(x, &scal[0]);
    k_finalize_mean<<<1, 1, 0, stream>>>(&scal[0], &scal[1], 1.0f / (float)N_DST0);

    // pack weights to f16 fragment-major
    {
        int th1 = F_IN * F_HID;   // 32768
        int th2 = F_HID * F_OUT;  // 8192
        k_pack_w<<<(th1 + 255) / 256, 256, 0, stream>>>(Wn1, w1n, F_HID, F_IN / 32, th1);
        k_pack_w<<<(th1 + 255) / 256, 256, 0, stream>>>(Ws1, w1s, F_HID, F_IN / 32, th1);
        k_pack_w<<<(th2 + 255) / 256, 256, 0, stream>>>(Wn2, w2n, F_OUT, F_HID / 32, th2);
        k_pack_w<<<(th2 + 255) / 256, 256, 0, stream>>>(Ws2, w2s, F_OUT, F_HID / 32, th2);
    }

    // layer 1 fused
    k_layer1<<<N_DST0 / 16, 256, 0, stream>>>(x, es0, w1n, w1s, b1, &scal[1], hbuf);

    // size_b_mean over h[:N_DST1]
    k_normsum_f16<<<N_DST1 / 8, 256, 0, stream>>>(hbuf, &scal[2]);
    k_finalize_mean<<<1, 1, 0, stream>>>(&scal[2], &scal[3], 1.0f / (float)N_DST1);

    // layer 2 fused
    k_layer2<<<N_DST1 / 32, 128, 0, stream>>>(hbuf, es1, w2n, w2s, b2, &scal[3], out);
}
